// MultiHeadAttention_41669772706263
// MI455X (gfx1250) — matrix operand order repaired and checked by reading, not compiled, not run on hardware
//
#include <hip/hip_runtime.h>

#define L_SEQ   2048
#define DMODEL  768
#define NHEAD   12
#define DK      64
#define NBATCH  4
#define NROWS   (NBATCH * L_SEQ)   // 8192
#define HB      (NHEAD * NBATCH)   // 48
#define NEG_FILL (-4294967295.0f)

typedef __attribute__((ext_vector_type(16))) __bf16 bf16x16;
typedef __attribute__((ext_vector_type(8)))  float  floatx8;

union FragCvt { uint4 u[2]; bf16x16 v; };

// 16 contiguous bf16 (B-matrix style fragment half-row)
__device__ __forceinline__ bf16x16 load_frag16(const __bf16* p) {
  FragCvt f;
  f.u[0] = *reinterpret_cast<const uint4*>(p);
  f.u[1] = *reinterpret_cast<const uint4*>(p + 8);
  return f.v;
}
// A-matrix fragment: 16 contiguous elems at p+2*aoff
__device__ __forceinline__ bf16x16 load_frag_split(const __bf16* p, int aoff) {
  FragCvt f;
  f.u[0] = *reinterpret_cast<const uint4*>(p + 2 * aoff);
  f.u[1] = *reinterpret_cast<const uint4*>(p + 2 * aoff + 8);
  return f.v;
}
__device__ __forceinline__ floatx8 wmma_bf16(bf16x16 a, bf16x16 b, floatx8 c) {
  return __builtin_amdgcn_wmma_f32_16x16x32_bf16(false, a, false, b, (short)0, c,
                                                 false, false);
}

// ---------------------------------------------------------------- conversion
__global__ __launch_bounds__(256)
void cvt_f32_bf16(const float* __restrict__ in, __bf16* __restrict__ out, int n) {
  int i = blockIdx.x * 256 + threadIdx.x;
  if (i < n) out[i] = (__bf16)in[i];
}

// ---------------------------------------------------------------- row masks
__global__ __launch_bounds__(256)
void row_sign_mask(const float* __restrict__ X, float* __restrict__ mask) {
  const int row = blockIdx.x;
  const int tid = threadIdx.x;
  const float* p = X + (size_t)row * DMODEL;
  __shared__ float rs[256];
  float s = 0.0f;
#pragma unroll
  for (int t = 0; t < 3; ++t) s += p[tid + t * 256];
  rs[tid] = s;
  __syncthreads();
  for (int off = 128; off > 0; off >>= 1) {
    if (tid < off) rs[tid] += rs[tid + off];
    __syncthreads();
  }
  if (tid == 0) mask[row] = (rs[0] != 0.0f) ? 1.0f : 0.0f;
}

// ------------------------------------------------- projection GEMM (NT, bf16)
// out = relu(X @ W^T + bias) * scale, scattered to head-split layout.
__device__ __forceinline__ void store_proj(__bf16* out, int row, int col, float v,
                                           int transposed) {
  int h = col >> 6, c = col & 63;
  int b = row >> 11, l = row & (L_SEQ - 1);
  size_t idx;
  if (transposed)  // V^T: [hb, d_k, L]
    idx = ((size_t)(h * NBATCH + b) * DK + c) * L_SEQ + l;
  else             // Q/K: [hb, L, d_k]
    idx = ((size_t)(h * NBATCH + b) * L_SEQ + l) * DK + c;
  out[idx] = (__bf16)v;
}

__global__ __launch_bounds__(128)
void gemm_bias_relu_proj(const __bf16* __restrict__ X,   // [8192, 768]
                         const __bf16* __restrict__ W,   // [768, 768] (row = out ch)
                         const float*  __restrict__ bias,
                         __bf16* __restrict__ out,
                         float scale, int transposed) {
  const int lane = threadIdx.x & 31;
  const int wave = threadIdx.x >> 5;
  const int m0 = blockIdx.x * 64 + (wave & 1) * 32;
  const int n0 = blockIdx.y * 64 + (wave >> 1) * 32;
  const int l16  = lane & 15;
  const int aoff = (lane < 16) ? 0 : 8;
  const int boff = (lane < 16) ? 0 : 16;

  const __bf16* pa0 = X + (size_t)(m0 + l16) * DMODEL;
  const __bf16* pa1 = X + (size_t)(m0 + 16 + l16) * DMODEL;
  const __bf16* pb0 = W + (size_t)(n0 + l16) * DMODEL + boff;
  const __bf16* pb1 = W + (size_t)(n0 + 16 + l16) * DMODEL + boff;

  floatx8 acc[2][2] = {};
  for (int k = 0; k < DMODEL; k += 32) {
    bf16x16 a0 = load_frag_split(pa0 + k, aoff);
    bf16x16 a1 = load_frag_split(pa1 + k, aoff);
    bf16x16 b0 = load_frag16(pb0 + k);
    bf16x16 b1 = load_frag16(pb1 + k);
    acc[0][0] = wmma_bf16(a0, b0, acc[0][0]);
    acc[0][1] = wmma_bf16(a0, b1, acc[0][1]);
    acc[1][0] = wmma_bf16(a1, b0, acc[1][0]);
    acc[1][1] = wmma_bf16(a1, b1, acc[1][1]);
  }

  const int rhalf = (lane >> 4) << 3;  // 0 or 8
#pragma unroll
  for (int tm = 0; tm < 2; ++tm)
#pragma unroll
    for (int tn = 0; tn < 2; ++tn) {
      int col = n0 + tn * 16 + l16;
      float bb = bias[col];
#pragma unroll
      for (int r = 0; r < 8; ++r) {
        int row = m0 + tm * 16 + rhalf + r;
        float v = acc[tm][tn][r] + bb;
        v = fmaxf(v, 0.0f) * scale;
        store_proj(out, row, col, v, transposed);
      }
    }
}

// ---------------------------------------------------- flash attention (bf16)
__global__ __launch_bounds__(128)
void flash_attention(const __bf16* __restrict__ Qh,   // [hb, L, 64], pre-scaled 1/8
                     const __bf16* __restrict__ Kh,   // [hb, L, 64]
                     const __bf16* __restrict__ Vt,   // [hb, 64, L]
                     const float* __restrict__ kmask, // [B, L]
                     const float* __restrict__ qmask, // [B, L]
                     float* __restrict__ attnout) {   // [B, L, 768]
  const int hb = blockIdx.x;
  const int h = hb / NBATCH;
  const int b = hb % NBATCH;
  const int lane = threadIdx.x & 31;
  const int wave = threadIdx.x >> 5;
  const int q0 = blockIdx.y * 64 + wave * 16;
  const int l16  = lane & 15;
  const int aoff = (lane < 16) ? 0 : 8;
  const int boff = (lane < 16) ? 0 : 16;
  const int rhalf = (lane >> 4) << 3;

  __shared__ __attribute__((aligned(16))) __bf16 pstage[4][16 * 32];
  __bf16* pst = pstage[wave];

  // Q A-fragments for this wave's 16 query rows (K = 64 -> two frags)
  const __bf16* qrow = Qh + ((size_t)hb * L_SEQ + q0 + l16) * DK;
  bf16x16 qa0 = load_frag_split(qrow, aoff);
  bf16x16 qa1 = load_frag_split(qrow + 32, aoff);

  floatx8 o0 = {}, o1 = {}, o2 = {}, o3 = {};
  float mrow[8], lrow[8];
#pragma unroll
  for (int r = 0; r < 8; ++r) { mrow[r] = -3.0e38f; lrow[r] = 0.0f; }

  const float*  km    = kmask + (size_t)b * L_SEQ;
  const __bf16* kbase = Kh + (size_t)hb * L_SEQ * DK;
  const __bf16* vbase = Vt + ((size_t)hb * DK + l16) * L_SEQ;

  for (int j = 0; j < L_SEQ; j += 32) {
    // --- S = Q . K^T for 32 keys (two 16-col tiles) ---
    const __bf16* kr0 = kbase + (size_t)(j + l16) * DK + boff;
    const __bf16* kr1 = kr0 + 16 * DK;
    bf16x16 kb00 = load_frag16(kr0);
    bf16x16 kb01 = load_frag16(kr0 + 32);
    bf16x16 kb10 = load_frag16(kr1);
    bf16x16 kb11 = load_frag16(kr1 + 32);

    floatx8 s0 = {}, s1 = {};
    s0 = wmma_bf16(qa0, kb00, s0);
    s0 = wmma_bf16(qa1, kb01, s0);
    s1 = wmma_bf16(qa0, kb10, s1);
    s1 = wmma_bf16(qa1, kb11, s1);

    if (j + 32 < L_SEQ) {  // global_prefetch_b8 of next K/V block
      __builtin_prefetch(kr0 + 32 * DK, 0, 3);
      __builtin_prefetch(vbase + j + 32, 0, 3);
    }

    float km0 = km[j + l16];
    float km1 = km[j + 16 + l16];

    // --- online softmax: per-row max/sum across 16 lanes of each half ---
#pragma unroll
    for (int r = 0; r < 8; ++r) {
      float e0 = (km0 == 0.0f) ? NEG_FILL : s0[r];
      float e1 = (km1 == 0.0f) ? NEG_FILL : s1[r];
      float mx = fmaxf(e0, e1);
      mx = fmaxf(mx, __shfl_xor(mx, 1, 32));
      mx = fmaxf(mx, __shfl_xor(mx, 2, 32));
      mx = fmaxf(mx, __shfl_xor(mx, 4, 32));
      mx = fmaxf(mx, __shfl_xor(mx, 8, 32));
      float mnew  = fmaxf(mrow[r], mx);
      float alpha = __expf(mrow[r] - mnew);
      float p0 = __expf(e0 - mnew);
      float p1 = __expf(e1 - mnew);
      float ps = p0 + p1;
      ps += __shfl_xor(ps, 1, 32);
      ps += __shfl_xor(ps, 2, 32);
      ps += __shfl_xor(ps, 4, 32);
      ps += __shfl_xor(ps, 8, 32);
      lrow[r] = lrow[r] * alpha + ps;
      mrow[r] = mnew;
      o0[r] *= alpha; o1[r] *= alpha; o2[r] *= alpha; o3[r] *= alpha;
      // stage P (C-layout -> memory row-major) through per-wave LDS
      int prow = rhalf + r;
      pst[prow * 32 + l16]      = (__bf16)p0;
      pst[prow * 32 + 16 + l16] = (__bf16)p1;
    }

    // --- O += P . V (A frag of P re-loaded in A layout from LDS) ---
    bf16x16 pfrag = load_frag_split(pst + l16 * 32, aoff);
    const __bf16* vb = vbase + j + boff;
    bf16x16 v0 = load_frag16(vb);
    bf16x16 v1 = load_frag16(vb + 16 * L_SEQ);
    bf16x16 v2 = load_frag16(vb + 32 * L_SEQ);
    bf16x16 v3 = load_frag16(vb + 48 * L_SEQ);
    o0 = wmma_bf16(pfrag, v0, o0);
    o1 = wmma_bf16(pfrag, v1, o1);
    o2 = wmma_bf16(pfrag, v2, o2);
    o3 = wmma_bf16(pfrag, v3, o3);
  }

  // --- finalize: softmax normalization, query mask, merge heads ---
#pragma unroll
  for (int r = 0; r < 8; ++r) {
    int qr = q0 + rhalf + r;
    float qm  = qmask[(size_t)b * L_SEQ + qr];
    float inv = qm / lrow[r];
    size_t base = ((size_t)b * L_SEQ + qr) * DMODEL + h * DK + l16;
    attnout[base]      = o0[r] * inv;
    attnout[base + 16] = o1[r] * inv;
    attnout[base + 32] = o2[r] * inv;
    attnout[base + 48] = o3[r] * inv;
  }
}

// ------------------------------------------------ residual + layernorm (f32)
__global__ __launch_bounds__(256)
void residual_layernorm(const float* __restrict__ attnout,
                        const float* __restrict__ query,
                        const float* __restrict__ gamma,
                        const float* __restrict__ beta,
                        float* __restrict__ out) {
  const int row = blockIdx.x;
  const int tid = threadIdx.x;
  const float* a = attnout + (size_t)row * DMODEL;
  const float* q = query   + (size_t)row * DMODEL;
  __shared__ float rs[256], rq[256];
  float x[3];
  float s = 0.0f, s2 = 0.0f;
#pragma unroll
  for (int t = 0; t < 3; ++t) {
    int i = tid + t * 256;
    float v = a[i] + q[i];
    x[t] = v; s += v; s2 += v * v;
  }
  rs[tid] = s; rq[tid] = s2;
  __syncthreads();
  for (int off = 128; off > 0; off >>= 1) {
    if (tid < off) { rs[tid] += rs[tid + off]; rq[tid] += rq[tid + off]; }
    __syncthreads();
  }
  float mean = rs[0] * (1.0f / DMODEL);
  float var  = (rq[0] - (float)DMODEL * mean * mean) * (1.0f / (DMODEL - 1)); // ddof=1
  float inv  = rsqrtf(var + 1e-5f);
#pragma unroll
  for (int t = 0; t < 3; ++t) {
    int i = tid + t * 256;
    out[(size_t)row * DMODEL + i] = gamma[i] * ((x[t] - mean) * inv) + beta[i];
  }
}

// ---------------------------------------------------------------- launcher
extern "C" void kernel_launch(void* const* d_in, const int* in_sizes, int n_in,
                              void* d_out, int out_size, void* d_ws, size_t ws_size,
                              hipStream_t stream) {
  const float* query = (const float*)d_in[0];
  const float* key   = (const float*)d_in[1];
  const float* Wq    = (const float*)d_in[2];
  const float* bq    = (const float*)d_in[3];
  const float* Wk    = (const float*)d_in[4];
  const float* bk    = (const float*)d_in[5];
  const float* Wv    = (const float*)d_in[6];
  const float* bv    = (const float*)d_in[7];
  const float* gamma = (const float*)d_in[8];
  const float* beta  = (const float*)d_in[9];
  float* out = (float*)d_out;

  char* ws = (char*)d_ws;
  size_t off = 0;
  auto carve = [&](size_t bytes) {
    void* p = ws + off;
    off += (bytes + 255) & ~(size_t)255;
    return p;
  };
  __bf16* xq_bf = (__bf16*)carve((size_t)NROWS * DMODEL * 2);
  __bf16* xk_bf = (__bf16*)carve((size_t)NROWS * DMODEL * 2);
  __bf16* wq_bf = (__bf16*)carve((size_t)DMODEL * DMODEL * 2);
  __bf16* wk_bf = (__bf16*)carve((size_t)DMODEL * DMODEL * 2);
  __bf16* wv_bf = (__bf16*)carve((size_t)DMODEL * DMODEL * 2);
  __bf16* Qh    = (__bf16*)carve((size_t)HB * L_SEQ * DK * 2);
  __bf16* Kh    = (__bf16*)carve((size_t)HB * L_SEQ * DK * 2);
  __bf16* Vt    = (__bf16*)carve((size_t)HB * DK * L_SEQ * 2);
  float* kmaskb = (float*)carve((size_t)NROWS * 4);
  float* qmaskb = (float*)carve((size_t)NROWS * 4);
  float* attnb  = (float*)carve((size_t)NROWS * DMODEL * 4);

  const int nX = NROWS * DMODEL;
  const int nW = DMODEL * DMODEL;
  cvt_f32_bf16<<<(nX + 255) / 256, 256, 0, stream>>>(query, xq_bf, nX);
  cvt_f32_bf16<<<(nX + 255) / 256, 256, 0, stream>>>(key,   xk_bf, nX);
  cvt_f32_bf16<<<(nW + 255) / 256, 256, 0, stream>>>(Wq, wq_bf, nW);
  cvt_f32_bf16<<<(nW + 255) / 256, 256, 0, stream>>>(Wk, wk_bf, nW);
  cvt_f32_bf16<<<(nW + 255) / 256, 256, 0, stream>>>(Wv, wv_bf, nW);

  row_sign_mask<<<NROWS, 256, 0, stream>>>(query, qmaskb);
  row_sign_mask<<<NROWS, 256, 0, stream>>>(key,   kmaskb);

  dim3 ggrid(NROWS / 64, DMODEL / 64);
  gemm_bias_relu_proj<<<ggrid, 128, 0, stream>>>(xq_bf, wq_bf, bq, Qh, 0.125f, 0);
  gemm_bias_relu_proj<<<ggrid, 128, 0, stream>>>(xk_bf, wk_bf, bk, Kh, 1.0f, 0);
  gemm_bias_relu_proj<<<ggrid, 128, 0, stream>>>(xk_bf, wv_bf, bv, Vt, 1.0f, 1);

  flash_attention<<<dim3(HB, L_SEQ / 64), 128, 0, stream>>>(Qh, Kh, Vt, kmaskb,
                                                            qmaskb, attnb);

  residual_layernorm<<<NROWS, 256, 0, stream>>>(attnb, query, gamma, beta, out);
}
